// EGNNEncoder_77008763617320
// MI455X (gfx1250) — matrix operand order repaired and checked by reading, not compiled, not run on hardware
//
#include <hip/hip_runtime.h>
#include <hip/hip_bf16.h>

// ---------------- problem constants (fixed by the reference) ----------------
#define NATOM   80000     // N
#define EBK     30000     // block edges
#define NEDGE   360000    // EB*U*K
#define HIDN    128
#define STR_IN  290       // LDS stride (ushorts) for 288-wide tile (odd dwords)
#define STR_IN2 258       // LDS stride for 256-wide tile
#define STR_H   130       // LDS stride for 128-wide tile

typedef __attribute__((ext_vector_type(16))) __bf16 v16bf;
typedef __attribute__((ext_vector_type(8)))  float  v8f;

union Frag { v16bf v; unsigned short u[16]; unsigned int d[8]; };

__device__ __forceinline__ unsigned short f2bf(float f) {
  unsigned int u = __float_as_uint(f);
  u += 0x7FFFu + ((u >> 16) & 1u);          // round-to-nearest-even
  return (unsigned short)(u >> 16);
}
__device__ __forceinline__ float bf2f(unsigned short s) {
  return __uint_as_float(((unsigned int)s) << 16);
}
__device__ __forceinline__ float silu_f(float x) { return x / (1.f + __expf(-x)); }

// A fragment: 16x32 bf16 row-major tile in LDS -> WMMA A layout.
// lane m (0..15) kh=0, lane m+16 kh=1; elem pairs: VGPR0-3 K=kh*8+{0..7},
// VGPR4-7 K=16+kh*8+{0..7}.
__device__ __forceinline__ void load_a_frag(const unsigned short* sA, int stride,
                                            int kt, int lane, Frag& a) {
  const int m = lane & 15, kh = (lane >> 4) & 1;
  const unsigned short* base = sA + m * stride + kt * 32 + kh * 8;
#pragma unroll
  for (int i = 0; i < 8; ++i) {
    const int kb = ((i & 3) << 1) + ((i & 4) << 2);   // 0,2,4,6,16,18,20,22
    a.u[2 * i]     = base[kb];
    a.u[2 * i + 1] = base[kb + 1];
  }
}

// B fragment: pre-packed in global so each lane loads 16 contiguous bf16.
__device__ __forceinline__ void load_b_frag(const unsigned short* wf, int kt,
                                            int nt, int lane, Frag& b) {
  const unsigned int* p =
      (const unsigned int*)(wf + ((((kt * 8 + nt) * 32) + lane) << 4));
#pragma unroll
  for (int i = 0; i < 8; ++i) b.d[i] = p[i];
}

__device__ __forceinline__ v8f wmma_bf16(const Frag& a, const Frag& b, v8f c) {
  return __builtin_amdgcn_wmma_f32_16x16x32_bf16(false, a.v, false, b.v,
                                                 (short)0, c, false, false);
}

// ------------- weight pre-pack: W[K,128] fp32 -> B-fragment bf16 -------------
__global__ __launch_bounds__(256) void pack_b_kernel(const float* __restrict__ W,
                                                     unsigned short* __restrict__ out,
                                                     int Kdim, int KP) {
  const int idx = blockIdx.x * blockDim.x + threadIdx.x;
  if (idx >= KP * 128) return;
  const int j    = idx & 15;
  const int lane = (idx >> 4) & 31;
  const int nt   = (idx >> 9) & 7;
  const int kt   = idx >> 12;
  const int nn = lane & 15, kh = lane >> 4;
  const int krow = kt * 32 + (j < 8 ? kh * 8 + j : 16 + kh * 8 + (j - 8));
  const int ncol = nt * 16 + nn;
  const float v = (krow < Kdim) ? W[(size_t)krow * 128 + ncol] : 0.f;
  out[idx] = f2bf(v);
}

// ---------------- misc small kernels ----------------
__global__ __launch_bounds__(256) void init_kernel(const float* __restrict__ Z,
                                                   float* __restrict__ x,
                                                   float* __restrict__ cnt) {
  const int i = blockIdx.x * blockDim.x + threadIdx.x;
  if (i < NATOM * 3) x[i] = Z[i];
  if (i < NATOM) cnt[i] = 0.f;
}
__global__ __launch_bounds__(256) void zero_kernel(float* __restrict__ p, int n) {
  const int i = blockIdx.x * blockDim.x + threadIdx.x;
  if (i < n) p[i] = 0.f;
}
__global__ __launch_bounds__(256) void copy_kernel(const float* __restrict__ a,
                                                   float* __restrict__ b, int n) {
  const int i = blockIdx.x * blockDim.x + threadIdx.x;
  if (i < n) b[i] = a[i];
}

// ---------------- edge construction (top-3 of 4, stable ties) ----------------
__global__ __launch_bounds__(256) void build_edges_kernel(const int* __restrict__ edges,
                                                          const float* __restrict__ Z,
                                                          int* __restrict__ row,
                                                          int* __restrict__ col,
                                                          float* __restrict__ cnt) {
  const int eb = blockIdx.x * blockDim.x + threadIdx.x;
  if (eb >= EBK) return;
  const int sb = edges[eb], db = edges[EBK + eb];
  float zs[4][3], zd[4][3];
#pragma unroll
  for (int u = 0; u < 4; ++u)
#pragma unroll
    for (int c = 0; c < 3; ++c) {
      zs[u][c] = Z[(size_t)(sb * 4 + u) * 3 + c];
      zd[u][c] = Z[(size_t)(db * 4 + u) * 3 + c];
    }
#pragma unroll
  for (int u = 0; u < 4; ++u) {
    float d2[4];
#pragma unroll
    for (int v = 0; v < 4; ++v) {
      const float dx = zs[u][0] - zd[v][0];
      const float dy = zs[u][1] - zd[v][1];
      const float dz = zs[u][2] - zd[v][2];
      d2[v] = dx * dx + dy * dy + dz * dz;
    }
    int used = 0;
#pragma unroll
    for (int k = 0; k < 3; ++k) {
      int best = 0; float bk = -3.0e38f;
#pragma unroll
      for (int v = 0; v < 4; ++v) {
        if (used & (1 << v)) continue;
        const float key = -d2[v];                 // top_k(-d2): strict >, first idx wins
        if (key > bk) { bk = key; best = v; }
      }
      used |= 1 << best;
      const int e = (eb * 4 + u) * 3 + k;
      row[e] = sb * 4 + u;
      col[e] = db * 4 + best;
    }
    atomicAdd(&cnt[sb * 4 + u], 3.0f);
  }
}

// ---------------- fused edge MLP: 16 edges / block, 4 waves ----------------
__global__ __launch_bounds__(128) void edge_mlp_kernel(
    const float* __restrict__ h, const float* __restrict__ x,
    const float* __restrict__ eattr, const int* __restrict__ row,
    const int* __restrict__ col,
    const unsigned short* __restrict__ w1f, const float* __restrict__ b1,
    const unsigned short* __restrict__ w2f, const float* __restrict__ b2,
    const unsigned short* __restrict__ cw1f, const float* __restrict__ cb1,
    const float* __restrict__ cw2,
    float* __restrict__ agg, float* __restrict__ xacc) {
  __shared__ unsigned short sIn[16 * STR_IN];
  __shared__ unsigned short sM1[16 * STR_H];
  __shared__ unsigned short sM2[16 * STR_H];
  __shared__ unsigned short sT[16 * STR_H];
  __shared__ float sDiff[16][3];
  __shared__ int   sRow[16];

  const int tid = threadIdx.x;
  const int e0  = blockIdx.x * 16;

  {  // gather inputs: [h_row(128) | h_col(128) | radial | eattr(16) | pad->288]
    const int el = tid >> 3, l8 = tid & 7;
    const int e = e0 + el;
    const int r = row[e], c = col[e];
    const float* hr = h + (size_t)r * HIDN + l8 * 16;
    const float* hc = h + (size_t)c * HIDN + l8 * 16;
    unsigned short* dst = sIn + el * STR_IN;
#pragma unroll
    for (int i = 0; i < 16; ++i) dst[l8 * 16 + i] = f2bf(hr[i]);
#pragma unroll
    for (int i = 0; i < 16; ++i) dst[HIDN + l8 * 16 + i] = f2bf(hc[i]);
    if (l8 == 0) {
      sRow[el] = r;
      const float dx = x[r * 3 + 0] - x[c * 3 + 0];
      const float dy = x[r * 3 + 1] - x[c * 3 + 1];
      const float dz = x[r * 3 + 2] - x[c * 3 + 2];
      sDiff[el][0] = dx; sDiff[el][1] = dy; sDiff[el][2] = dz;
      dst[256] = f2bf(dx * dx + dy * dy + dz * dz);
      const float* ea = eattr + (size_t)(e / 12) * 16;   // edge_id = e / (U*K)
#pragma unroll
      for (int i = 0; i < 16; ++i) dst[257 + i] = f2bf(ea[i]);
#pragma unroll
      for (int i = 273; i < 288; ++i) dst[i] = 0;
    }
  }
  __syncthreads();
  const int lane = tid & 31, wave = tid >> 5;
  const int n = lane & 15, kh = lane >> 4;
  const int nt0 = wave * 2, nt1 = nt0 + 1;

  {  // GEMM1: [16,288]@[288,128] + bias + silu -> sM1
    v8f a0 = {}, a1 = {};
    for (int kt = 0; kt < 9; ++kt) {
      Frag af; load_a_frag(sIn, STR_IN, kt, lane, af);
      Frag bf0; load_b_frag(w1f, kt, nt0, lane, bf0);
      Frag bf1; load_b_frag(w1f, kt, nt1, lane, bf1);
      a0 = wmma_bf16(af, bf0, a0);
      a1 = wmma_bf16(af, bf1, a1);
    }
    const float bc0 = b1[nt0 * 16 + n], bc1 = b1[nt1 * 16 + n];
#pragma unroll
    for (int r = 0; r < 8; ++r) {
      sM1[(r + 8 * kh) * STR_H + nt0 * 16 + n] = f2bf(silu_f(a0[r] + bc0));
      sM1[(r + 8 * kh) * STR_H + nt1 * 16 + n] = f2bf(silu_f(a1[r] + bc1));
    }
  }
  __syncthreads();
  {  // GEMM2: [16,128]@[128,128] + bias + silu -> sM2 + atomic segment-sum
    v8f a0 = {}, a1 = {};
    for (int kt = 0; kt < 4; ++kt) {
      Frag af; load_a_frag(sM1, STR_H, kt, lane, af);
      Frag bf0; load_b_frag(w2f, kt, nt0, lane, bf0);
      Frag bf1; load_b_frag(w2f, kt, nt1, lane, bf1);
      a0 = wmma_bf16(af, bf0, a0);
      a1 = wmma_bf16(af, bf1, a1);
    }
    const float bc0 = b2[nt0 * 16 + n], bc1 = b2[nt1 * 16 + n];
#pragma unroll
    for (int r = 0; r < 8; ++r) {
      const int m = r + 8 * kh;
      const float v0 = silu_f(a0[r] + bc0);
      const float v1 = silu_f(a1[r] + bc1);
      sM2[m * STR_H + nt0 * 16 + n] = f2bf(v0);
      sM2[m * STR_H + nt1 * 16 + n] = f2bf(v1);
      atomicAdd(&agg[(size_t)sRow[m] * HIDN + nt0 * 16 + n], v0);
      atomicAdd(&agg[(size_t)sRow[m] * HIDN + nt1 * 16 + n], v1);
    }
  }
  __syncthreads();
  {  // GEMM3 (coord MLP1): silu(m@cw1 + cb1) -> sT
    v8f a0 = {}, a1 = {};
    for (int kt = 0; kt < 4; ++kt) {
      Frag af; load_a_frag(sM2, STR_H, kt, lane, af);
      Frag bf0; load_b_frag(cw1f, kt, nt0, lane, bf0);
      Frag bf1; load_b_frag(cw1f, kt, nt1, lane, bf1);
      a0 = wmma_bf16(af, bf0, a0);
      a1 = wmma_bf16(af, bf1, a1);
    }
    const float bc0 = cb1[nt0 * 16 + n], bc1 = cb1[nt1 * 16 + n];
#pragma unroll
    for (int r = 0; r < 8; ++r) {
      sT[(r + 8 * kh) * STR_H + nt0 * 16 + n] = f2bf(silu_f(a0[r] + bc0));
      sT[(r + 8 * kh) * STR_H + nt1 * 16 + n] = f2bf(silu_f(a1[r] + bc1));
    }
  }
  __syncthreads();
  if (tid < 16) {  // w = t @ coord_w2 ; scatter diff*w
    const unsigned short* t = sT + tid * STR_H;
    float w = 0.f;
#pragma unroll 4
    for (int j = 0; j < HIDN; ++j) w += bf2f(t[j]) * cw2[j];
    const int r = sRow[tid];
    atomicAdd(&xacc[r * 3 + 0], sDiff[tid][0] * w);
    atomicAdd(&xacc[r * 3 + 1], sDiff[tid][1] * w);
    atomicAdd(&xacc[r * 3 + 2], sDiff[tid][2] * w);
  }
}

// -------- node update (fused x update): h += silu([h|agg]W1+b1)W2 + b2 --------
__global__ __launch_bounds__(128) void node_kernel(
    float* __restrict__ h, const float* __restrict__ agg,
    float* __restrict__ x, const float* __restrict__ xacc,
    const float* __restrict__ cnt,
    const unsigned short* __restrict__ w1f, const float* __restrict__ b1,
    const unsigned short* __restrict__ w2f, const float* __restrict__ b2) {
  __shared__ unsigned short sIn[16 * STR_IN2];
  __shared__ unsigned short sM1[16 * STR_H];
  const int tid = threadIdx.x;
  const int n0 = blockIdx.x * 16;
  {
    const int el = tid >> 3, l8 = tid & 7;
    const size_t nd = (size_t)(n0 + el);
    const float* hp = h + nd * HIDN + l8 * 16;
    const float* ap = agg + nd * HIDN + l8 * 16;
    unsigned short* dst = sIn + el * STR_IN2;
#pragma unroll
    for (int i = 0; i < 16; ++i) dst[l8 * 16 + i] = f2bf(hp[i]);
#pragma unroll
    for (int i = 0; i < 16; ++i) dst[HIDN + l8 * 16 + i] = f2bf(ap[i]);
  }
  if (tid < 48) {  // x += xacc / max(cnt, 1)
    const int nn = tid / 3, cc = tid % 3;
    const int idx = (n0 + nn) * 3 + cc;
    x[idx] += xacc[idx] / fmaxf(cnt[n0 + nn], 1.f);
  }
  __syncthreads();
  const int lane = tid & 31, wave = tid >> 5;
  const int n = lane & 15, kh = lane >> 4;
  const int nt0 = wave * 2, nt1 = nt0 + 1;
  {
    v8f a0 = {}, a1 = {};
    for (int kt = 0; kt < 8; ++kt) {
      Frag af; load_a_frag(sIn, STR_IN2, kt, lane, af);
      Frag bf0; load_b_frag(w1f, kt, nt0, lane, bf0);
      Frag bf1; load_b_frag(w1f, kt, nt1, lane, bf1);
      a0 = wmma_bf16(af, bf0, a0);
      a1 = wmma_bf16(af, bf1, a1);
    }
    const float bc0 = b1[nt0 * 16 + n], bc1 = b1[nt1 * 16 + n];
#pragma unroll
    for (int r = 0; r < 8; ++r) {
      sM1[(r + 8 * kh) * STR_H + nt0 * 16 + n] = f2bf(silu_f(a0[r] + bc0));
      sM1[(r + 8 * kh) * STR_H + nt1 * 16 + n] = f2bf(silu_f(a1[r] + bc1));
    }
  }
  __syncthreads();
  {
    v8f a0 = {}, a1 = {};
    for (int kt = 0; kt < 4; ++kt) {
      Frag af; load_a_frag(sM1, STR_H, kt, lane, af);
      Frag bf0; load_b_frag(w2f, kt, nt0, lane, bf0);
      Frag bf1; load_b_frag(w2f, kt, nt1, lane, bf1);
      a0 = wmma_bf16(af, bf0, a0);
      a1 = wmma_bf16(af, bf1, a1);
    }
    const float bc0 = b2[nt0 * 16 + n], bc1 = b2[nt1 * 16 + n];
#pragma unroll
    for (int r = 0; r < 8; ++r) {
      const size_t m = (size_t)(n0 + r + 8 * kh);
      h[m * HIDN + nt0 * 16 + n] += a0[r] + bc0;   // residual
      h[m * HIDN + nt1 * 16 + n] += a1[r] + bc1;
    }
  }
}

// -------- plain GEMM+bias (emb_in / emb_out): out = A[.,128]@W + b --------
__global__ __launch_bounds__(128) void emb_kernel(const float* __restrict__ A,
                                                  const unsigned short* __restrict__ wf,
                                                  const float* __restrict__ bias,
                                                  float* __restrict__ out) {
  __shared__ unsigned short sIn[16 * STR_H];
  const int tid = threadIdx.x;
  const int r0 = blockIdx.x * 16;
  {
    const int el = tid >> 3, l8 = tid & 7;
    const float* a = A + (size_t)(r0 + el) * HIDN + l8 * 16;
    unsigned short* dst = sIn + el * STR_H;
#pragma unroll
    for (int i = 0; i < 16; ++i) dst[l8 * 16 + i] = f2bf(a[i]);
  }
  __syncthreads();
  const int lane = tid & 31, wave = tid >> 5;
  const int n = lane & 15, kh = lane >> 4;
  const int nt0 = wave * 2, nt1 = nt0 + 1;
  v8f a0 = {}, a1 = {};
  for (int kt = 0; kt < 4; ++kt) {
    Frag af; load_a_frag(sIn, STR_H, kt, lane, af);
    Frag bf0; load_b_frag(wf, kt, nt0, lane, bf0);
    Frag bf1; load_b_frag(wf, kt, nt1, lane, bf1);
    a0 = wmma_bf16(af, bf0, a0);
    a1 = wmma_bf16(af, bf1, a1);
  }
  const float bc0 = bias[nt0 * 16 + n], bc1 = bias[nt1 * 16 + n];
#pragma unroll
  for (int r = 0; r < 8; ++r) {
    const size_t m = (size_t)(r0 + r + 8 * kh);
    out[m * HIDN + nt0 * 16 + n] = a0[r] + bc0;
    out[m * HIDN + nt1 * 16 + n] = a1[r] + bc1;
  }
}

// ---------------- host orchestration ----------------
extern "C" void kernel_launch(void* const* d_in, const int* in_sizes, int n_in,
                              void* d_out, int out_size, void* d_ws, size_t ws_size,
                              hipStream_t stream) {
  (void)in_sizes; (void)n_in; (void)out_size; (void)ws_size;
  const float* H         = (const float*)d_in[0];
  const float* Z         = (const float*)d_in[1];
  const int*   edges     = (const int*)d_in[4];
  const float* edge_attr = (const float*)d_in[5];
  const float* emb_in_w  = (const float*)d_in[6];
  const float* emb_in_b  = (const float*)d_in[7];
  const float* emb_out_w = (const float*)d_in[8];
  const float* emb_out_b = (const float*)d_in[9];
  const float* edge_w1   = (const float*)d_in[10];
  const float* edge_b1   = (const float*)d_in[11];
  const float* edge_w2   = (const float*)d_in[12];
  const float* edge_b2   = (const float*)d_in[13];
  const float* node_w1   = (const float*)d_in[14];
  const float* node_b1   = (const float*)d_in[15];
  const float* node_w2   = (const float*)d_in[16];
  const float* node_b2   = (const float*)d_in[17];
  const float* coord_w1  = (const float*)d_in[18];
  const float* coord_b1  = (const float*)d_in[19];
  const float* coord_w2  = (const float*)d_in[20];

  char* ws = (char*)d_ws;
  size_t off = 0;
  auto salloc = [&](size_t bytes) -> char* {
    char* p = ws + off;
    off = (off + bytes + 255) & ~(size_t)255;
    return p;
  };
  int*   rowp = (int*)salloc((size_t)NEDGE * 4);
  int*   colp = (int*)salloc((size_t)NEDGE * 4);
  float* hbuf = (float*)salloc((size_t)NATOM * HIDN * 4);
  float* xbuf = (float*)salloc((size_t)NATOM * 3 * 4);
  float* aggb = (float*)salloc((size_t)NATOM * HIDN * 4);
  float* xacc = (float*)salloc((size_t)NATOM * 3 * 4);
  float* cntb = (float*)salloc((size_t)NATOM * 4);
  unsigned short* wfEmbIn  = (unsigned short*)salloc(128 * 128 * 2);
  unsigned short* wfEmbOut = (unsigned short*)salloc(128 * 128 * 2);
  unsigned short* wfE1[3]; unsigned short* wfE2[3];
  unsigned short* wfN1[3]; unsigned short* wfN2[3]; unsigned short* wfC1[3];
  for (int l = 0; l < 3; ++l) {
    wfE1[l] = (unsigned short*)salloc(288 * 128 * 2);
    wfE2[l] = (unsigned short*)salloc(128 * 128 * 2);
    wfN1[l] = (unsigned short*)salloc(256 * 128 * 2);
    wfN2[l] = (unsigned short*)salloc(128 * 128 * 2);
    wfC1[l] = (unsigned short*)salloc(128 * 128 * 2);
  }

  auto pack = [&](const float* W, unsigned short* o, int Kdim, int KP) {
    const int total = KP * 128;
    pack_b_kernel<<<(total + 255) / 256, 256, 0, stream>>>(W, o, Kdim, KP);
  };
  pack(emb_in_w, wfEmbIn, 128, 128);
  pack(emb_out_w, wfEmbOut, 128, 128);
  for (int l = 0; l < 3; ++l) {
    pack(edge_w1 + (size_t)l * 273 * 128, wfE1[l], 273, 288);
    pack(edge_w2 + (size_t)l * 128 * 128, wfE2[l], 128, 128);
    pack(node_w1 + (size_t)l * 256 * 128, wfN1[l], 256, 256);
    pack(node_w2 + (size_t)l * 128 * 128, wfN2[l], 128, 128);
    pack(coord_w1 + (size_t)l * 128 * 128, wfC1[l], 128, 128);
  }

  init_kernel<<<(NATOM * 3 + 255) / 256, 256, 0, stream>>>(Z, xbuf, cntb);
  build_edges_kernel<<<(EBK + 255) / 256, 256, 0, stream>>>(edges, Z, rowp, colp, cntb);
  emb_kernel<<<NATOM / 16, 128, 0, stream>>>(H, wfEmbIn, emb_in_b, hbuf);

  for (int l = 0; l < 3; ++l) {
    zero_kernel<<<(NATOM * HIDN + 255) / 256, 256, 0, stream>>>(aggb, NATOM * HIDN);
    zero_kernel<<<(NATOM * 3 + 255) / 256, 256, 0, stream>>>(xacc, NATOM * 3);
    edge_mlp_kernel<<<NEDGE / 16, 128, 0, stream>>>(
        hbuf, xbuf, edge_attr, rowp, colp,
        wfE1[l], edge_b1 + l * 128,
        wfE2[l], edge_b2 + l * 128,
        wfC1[l], coord_b1 + l * 128, coord_w2 + l * 128,
        aggb, xacc);
    node_kernel<<<NATOM / 16, 128, 0, stream>>>(
        hbuf, aggb, xbuf, xacc, cntb,
        wfN1[l], node_b1 + l * 128, wfN2[l], node_b2 + l * 128);
  }

  emb_kernel<<<NATOM / 16, 128, 0, stream>>>(hbuf, wfEmbOut, emb_out_b, (float*)d_out);
  copy_kernel<<<(NATOM * 3 + 255) / 256, 256, 0, stream>>>(
      xbuf, (float*)d_out + (size_t)NATOM * HIDN, NATOM * 3);
}
